// FusedUpsampleAdaFM_34617436406292
// MI455X (gfx1250) — matrix operand order repaired and checked by reading, not compiled, not run on hardware
//
#include <hip/hip_runtime.h>
#include <hip/hip_bf16.h>

typedef __attribute__((ext_vector_type(16))) __bf16 v16bf;
typedef __attribute__((ext_vector_type(8)))  __bf16 v8bf;
typedef __attribute__((ext_vector_type(8)))  float  v8f;

namespace {
constexpr int N_BATCH = 32;
constexpr int IC = 512;
constexpr int OC = 256;
constexpr int H  = 32;
constexpr int OH = 64;
constexpr int TD = 128;
constexpr float S1 = 0.125f;                  // sqrt(2/128)  (EqualLinear scale)
constexpr float S2 = 0.04419417382415922f;    // sqrt(2/(out_ch*kh)) = sqrt(2/1024)

__device__ __forceinline__ __bf16 to_bf16(float f) {
  unsigned u = __builtin_bit_cast(unsigned, f);
  u += 0x7FFFu + ((u >> 16) & 1u);            // round-to-nearest-even
  unsigned short h = (unsigned short)(u >> 16);
  return __builtin_bit_cast(__bf16, h);
}
} // namespace

// ---------------------------------------------------------------------------
// Kernel 1: 4-tap smoothed kernel  wk[ki][kj][ic][oc]  (tap-major, oc coalesced)
// ---------------------------------------------------------------------------
__global__ __launch_bounds__(256) void k_wkprep(const float* __restrict__ W,
                                                float* __restrict__ wk_t) {
  int idx = blockIdx.x * 256 + threadIdx.x;   // ic*256 + oc
  if (idx >= IC * OC) return;
  float w[3][3];
  const float* p = W + (size_t)idx * 9;
#pragma unroll
  for (int u = 0; u < 3; ++u)
#pragma unroll
    for (int v = 0; v < 3; ++v) w[u][v] = p[u * 3 + v];
#pragma unroll
  for (int i = 0; i < 4; ++i) {
#pragma unroll
    for (int j = 0; j < 4; ++j) {
      float s = 0.f;
      if (i > 0 && j > 0) s += w[i - 1][j - 1];
      if (i > 0 && j < 3) s += w[i - 1][j];
      if (i < 3 && j > 0) s += w[i][j - 1];
      if (i < 3 && j < 3) s += w[i][j];
      wk_t[(size_t)(i * 4 + j) * (IC * OC) + idx] = 0.25f * s;
    }
  }
}

// ---------------------------------------------------------------------------
// Kernel 2: gamma/beta GEMM via bf16 WMMA.  M=262144 rows, N=32, K=128.
// ---------------------------------------------------------------------------
__global__ __launch_bounds__(256) void k_gammabeta(const float* __restrict__ task,
                                                   const float* __restrict__ task_w,
                                                   const float* __restrict__ task_b,
                                                   float* __restrict__ gamma,
                                                   float* __restrict__ beta) {
  __shared__ __align__(16) __bf16 lA[128 * 40];   // 128 rows x 32 k (chunked)
  __shared__ __align__(16) __bf16 lB[32 * 144];   // 32 n x 128 k
  const int tid  = threadIdx.x;
  const int lane = tid & 31, wv = tid >> 5;
  const int sel  = lane >> 4, lrow = lane & 15;
  const int row0 = blockIdx.x * 128;

  for (int i = tid; i < 32 * 128; i += 256) {
    int n = i >> 7, k = i & 127;
    lB[n * 144 + k] = to_bf16(task[n * 128 + k]);
  }

  v8f acc[2] = {};
  for (int kc = 0; kc < 4; ++kc) {
    __syncthreads();
    {
      const int r = tid & 127, half = tid >> 7;
      const float* src = task_w + (size_t)(row0 + r) * TD + kc * 32;
#pragma unroll
      for (int it = 0; it < 16; ++it) {
        const int k = half + it * 2;
        lA[r * 40 + k] = to_bf16(src[k]);
      }
    }
    __syncthreads();
    v16bf af;
    {
      const __bf16* p = &lA[(wv * 16 + lrow) * 40];
      v8bf p0 = *(const v8bf*)(p + sel * 8);
      v8bf p1 = *(const v8bf*)(p + 16 + sel * 8);
#pragma unroll
      for (int i = 0; i < 8; ++i) { af[i] = p0[i]; af[i + 8] = p1[i]; }
    }
#pragma unroll
    for (int ns = 0; ns < 2; ++ns) {
      const __bf16* p = &lB[(ns * 16 + lrow) * 144 + kc * 32];
      v8bf p0 = *(const v8bf*)(p + sel * 8);
      v8bf p1 = *(const v8bf*)(p + 16 + sel * 8);
      v16bf bfr;
#pragma unroll
      for (int i = 0; i < 8; ++i) { bfr[i] = p0[i]; bfr[i + 8] = p1[i]; }
      acc[ns] = __builtin_amdgcn_wmma_f32_16x16x32_bf16(
          false, af, false, bfr, (short)0, acc[ns], false, false);
    }
  }
#pragma unroll
  for (int ns = 0; ns < 2; ++ns) {
    const int n = ns * 16 + lrow;
#pragma unroll
    for (int v = 0; v < 8; ++v) {
      const int row = row0 + wv * 16 + v + sel * 8;
      const float val = acc[ns][v] * S1 + task_b[row];
      const int j = row >> 1;
      if (row & 1) beta[(size_t)n * (IC * OC) + j]  = val;
      else         gamma[(size_t)n * (IC * OC) + j] = val;
    }
  }
}

// ---------------------------------------------------------------------------
// Kernel 3: per-sample bias  bi[n][oc]
// ---------------------------------------------------------------------------
__global__ __launch_bounds__(256) void k_bias(const float* __restrict__ task,
                                              const float* __restrict__ tb_w,
                                              const float* __restrict__ tb_b,
                                              const float* __restrict__ b,
                                              float* __restrict__ bi) {
  int idx = blockIdx.x * 256 + threadIdx.x;
  if (idx >= N_BATCH * OC) return;
  const int n = idx >> 8, oc = idx & 255;
  const float* t = task + n * TD;
  const float* w = tb_w + oc * TD;
  float s = 0.f;
  for (int k = 0; k < TD; ++k) s = fmaf(t[k], w[k], s);
  bi[idx] = b[oc] + tb_b[oc] + s * S1;
}

// ---------------------------------------------------------------------------
// Kernel 4 (fast-path prep): x[n][ic][h][w] f32 -> xbf[n][h][w][ic] bf16
// LDS-tiled transpose; odd-stride padding (65) keeps both phases conflict-free.
// ---------------------------------------------------------------------------
__global__ __launch_bounds__(256) void k_xprep(const float* __restrict__ x,
                                               __bf16* __restrict__ xbf) {
  __shared__ float tile[32][65];   // [w][icl]
  int bid = blockIdx.x;
  const int h = bid & 31, n = bid >> 5;
  const int tid = threadIdx.x;
  for (int t8 = 0; t8 < 8; ++t8) {   // 8 tiles of 64 ic
    __syncthreads();
    {
      const int w = tid & 31, ich = tid >> 5;          // ich 0..7
#pragma unroll
      for (int i = 0; i < 8; ++i) {
        const int icl = ich + i * 8;                    // 0..63
        const int ic = t8 * 64 + icl;
        tile[w][icl] = x[(((size_t)n * IC + ic) * H + h) * H + w];
      }
    }
    __syncthreads();
    {
      const int icl = tid & 63, wh = tid >> 6;          // wh 0..3
#pragma unroll
      for (int i = 0; i < 8; ++i) {
        const int w = wh + i * 4;
        xbf[((size_t)(n * H + h) * H + w) * IC + t8 * 64 + icl] = to_bf16(tile[w][icl]);
      }
    }
  }
}

// ---------------------------------------------------------------------------
// Kernel 5 (fast-path prep): modulated bf16 weights
//   Wmod[n][p=ki*4+kj][oc][ic] = bf16((wk_t[p][ic][oc]*gamma+beta)*S2)
// Transposed (ic-contiguous) so conv A-tiles are straight b128 copies.
// ---------------------------------------------------------------------------
__global__ __launch_bounds__(256) void k_wmod(const float* __restrict__ wk_t,
                                              const float* __restrict__ gamma,
                                              const float* __restrict__ beta,
                                              __bf16* __restrict__ wmod) {
  __shared__ float tile[64][65];   // [icl][ocl]
  int bid = blockIdx.x;
  const int ot = bid & 3;  bid >>= 2;   // oc tile (4 x 64)
  const int it = bid & 7;  bid >>= 3;   // ic tile (8 x 64)
  const int p  = bid & 15; bid >>= 4;   // plane
  const int n  = bid;                   // sample
  const int tid = threadIdx.x;
  const float* wkp = wk_t + (size_t)p * (IC * OC);
  {
    const int ocl = tid & 63, ich = tid >> 6;   // ich 0..3
#pragma unroll
    for (int i = 0; i < 16; ++i) {
      const int icl = ich + i * 4;
      const int ic = it * 64 + icl, oc = ot * 64 + ocl;
      const int gi = (n * IC + ic) * OC + oc;
      tile[icl][ocl] = fmaf(wkp[ic * OC + oc], gamma[gi], beta[gi]) * S2;
    }
  }
  __syncthreads();
  {
    const int icl = tid & 63, och = tid >> 6;   // och 0..3
#pragma unroll
    for (int i = 0; i < 16; ++i) {
      const int ocl = och + i * 4;
      const int oc = ot * 64 + ocl, ic = it * 64 + icl;
      wmod[((size_t)(n * 16 + p) * OC + oc) * IC + ic] = to_bf16(tile[icl][ocl]);
    }
  }
}

// ---------------------------------------------------------------------------
// Kernel 6: FAST conv. All staging is 16B vector copies of pre-packed bf16.
// Per thread per K-chunk: 4 global b128 + 4 ds_store_b128; 12 ds_load_b128 +
// 8 WMMA per wave.
// ---------------------------------------------------------------------------
__global__ __launch_bounds__(256) void k_conv_fast(const __bf16* __restrict__ xbf,
                                                   const __bf16* __restrict__ wmod,
                                                   const float* __restrict__ bi,
                                                   float* __restrict__ out) {
  __shared__ __align__(16) __bf16 lA[128 * 40];   // [ocl][icl]
  __shared__ __align__(16) __bf16 lB[128 * 40];   // [pix][icl]

  int bid = blockIdx.x;
  const int nb = bid & 7; bid >>= 3;
  const int mb = bid & 1; bid >>= 1;
  const int ph = bid & 1; bid >>= 1;
  const int pw = bid & 1; bid >>= 1;
  const int n  = bid;

  const int tid  = threadIdx.x;
  const int lane = tid & 31, wv = tid >> 5;
  const int sel  = lane >> 4, lrow = lane & 15;
  const int wm   = wv >> 1,  wn = wv & 1;
  const int q    = tid & 1;              // 32B half of a 64B row
  const int row  = tid >> 1;             // 0..127: A row (ocl) & B pixel row

  v8f acc[2][4] = {};

  for (int tap = 0; tap < 4; ++tap) {
    const int a = tap >> 1, bc = tap & 1;
    const int dr = a - 1 + ph,  dc = bc - 1 + pw;
    const int ki = 3 - (2 * a + ph), kj = 3 - (2 * bc + pw);
    const int plane = ki * 4 + kj;

    const __bf16* wsrc = wmod + ((size_t)(n * 16 + plane) * OC + mb * 128 + row) * IC;
    const int rr = row >> 5, c = row & 31;
    const int hr = nb * 4 + rr + dr, wc = c + dc;
    const bool ok = ((unsigned)hr < 32u) && ((unsigned)wc < 32u);
    const __bf16* bsrc = xbf + ((size_t)(n * H + (ok ? hr : 0)) * H + (ok ? wc : 0)) * IC;

    for (int ics = 0; ics < IC; ics += 32) {
      __syncthreads();
      { // A tile: straight b128 copy
        const uint4* g = (const uint4*)(wsrc + ics + q * 16);
        uint4* d = (uint4*)&lA[row * 40 + q * 16];
        d[0] = g[0]; d[1] = g[1];
      }
      { // B tile: b128 copy, zero for border rows
        uint4* d = (uint4*)&lB[row * 40 + q * 16];
        if (ok) {
          const uint4* g = (const uint4*)(bsrc + ics + q * 16);
          d[0] = g[0]; d[1] = g[1];
        } else {
          const uint4 z = {0u, 0u, 0u, 0u};
          d[0] = z; d[1] = z;
        }
      }
      __syncthreads();

      v16bf af[2], bfr[4];
#pragma unroll
      for (int ms = 0; ms < 2; ++ms) {
        const __bf16* p = &lA[(wm * 32 + ms * 16 + lrow) * 40];
        v8bf p0 = *(const v8bf*)(p + sel * 8);
        v8bf p1 = *(const v8bf*)(p + 16 + sel * 8);
#pragma unroll
        for (int i = 0; i < 8; ++i) { af[ms][i] = p0[i]; af[ms][i + 8] = p1[i]; }
      }
#pragma unroll
      for (int ns = 0; ns < 4; ++ns) {
        const __bf16* p = &lB[(wn * 64 + ns * 16 + lrow) * 40];
        v8bf p0 = *(const v8bf*)(p + sel * 8);
        v8bf p1 = *(const v8bf*)(p + 16 + sel * 8);
#pragma unroll
        for (int i = 0; i < 8; ++i) { bfr[ns][i] = p0[i]; bfr[ns][i + 8] = p1[i]; }
      }
#pragma unroll
      for (int ms = 0; ms < 2; ++ms)
#pragma unroll
        for (int ns = 0; ns < 4; ++ns)
          acc[ms][ns] = __builtin_amdgcn_wmma_f32_16x16x32_bf16(
              false, af[ms], false, bfr[ns], (short)0, acc[ms][ns], false, false);
    }
  }

#pragma unroll
  for (int ms = 0; ms < 2; ++ms) {
    const int m0 = mb * 128 + wm * 32 + ms * 16;
#pragma unroll
    for (int ns = 0; ns < 4; ++ns) {
      const int pix = nb * 128 + wn * 64 + ns * 16 + lrow;
      const int r = pix >> 5, c = pix & 31;
      const int oh = 2 * r + ph, ow = 2 * c + pw;
#pragma unroll
      for (int v = 0; v < 8; ++v) {
        const int oc = m0 + v + sel * 8;
        out[(((size_t)n * OC + oc) * OH + oh) * OH + ow] =
            acc[ms][ns][v] + bi[n * OC + oc];
      }
    }
  }
}

// ---------------------------------------------------------------------------
// Kernel 7: FALLBACK conv (round-1 fused modulation) — used if ws too small.
// ---------------------------------------------------------------------------
__global__ __launch_bounds__(256) void k_conv(const float* __restrict__ x,
                                              const float* __restrict__ wk_t,
                                              const float* __restrict__ gamma,
                                              const float* __restrict__ beta,
                                              const float* __restrict__ bi,
                                              float* __restrict__ out) {
  __shared__ __align__(16) __bf16 lA[128 * 40];
  __shared__ __align__(16) __bf16 lB[128 * 40];

  int bid = blockIdx.x;
  const int nb = bid & 7; bid >>= 3;
  const int mb = bid & 1; bid >>= 1;
  const int ph = bid & 1; bid >>= 1;
  const int pw = bid & 1; bid >>= 1;
  const int n  = bid;

  const int tid  = threadIdx.x;
  const int lane = tid & 31, wv = tid >> 5;
  const int sel  = lane >> 4, lrow = lane & 15;
  const int wm   = wv >> 1,  wn = wv & 1;

  v8f acc[2][4] = {};

  for (int tap = 0; tap < 4; ++tap) {
    const int a = tap >> 1, bc = tap & 1;
    const int dr = a - 1 + ph,  dc = bc - 1 + pw;
    const int ki = 3 - (2 * a + ph), kj = 3 - (2 * bc + pw);
    const float* wkp = wk_t + (size_t)(ki * 4 + kj) * (IC * OC);

    for (int ics = 0; ics < IC; ics += 32) {
      __syncthreads();
      {
        const int ocl = tid & 127, half = tid >> 7;
        const int oc = mb * 128 + ocl;
#pragma unroll
        for (int it = 0; it < 16; ++it) {
          const int icl = half + it * 2;
          const int ic  = ics + icl;
          const int gi  = (n * IC + ic) * OC + oc;
          const float av = fmaf(wkp[ic * OC + oc], gamma[gi], beta[gi]) * S2;
          lA[ocl * 40 + icl] = to_bf16(av);
        }
      }
      {
        const int c = tid & 31, g8 = tid >> 5;
        const int wc = c + dc;
        const bool cok = (unsigned)wc < 32u;
#pragma unroll
        for (int it = 0; it < 16; ++it) {
          const int comb = it * 8 + g8;
          const int icl = comb & 31, rr = comb >> 5;
          const int hr = nb * 4 + rr + dr;
          float v = 0.f;
          if (cok && (unsigned)hr < 32u)
            v = x[(((size_t)n * IC + (ics + icl)) * H + hr) * H + wc];
          lB[(rr * 32 + c) * 40 + icl] = to_bf16(v);
        }
      }
      __syncthreads();

      v16bf af[2], bfr[4];
#pragma unroll
      for (int ms = 0; ms < 2; ++ms) {
        const __bf16* p = &lA[(wm * 32 + ms * 16 + lrow) * 40];
        v8bf p0 = *(const v8bf*)(p + sel * 8);
        v8bf p1 = *(const v8bf*)(p + 16 + sel * 8);
#pragma unroll
        for (int i = 0; i < 8; ++i) { af[ms][i] = p0[i]; af[ms][i + 8] = p1[i]; }
      }
#pragma unroll
      for (int ns = 0; ns < 4; ++ns) {
        const __bf16* p = &lB[(wn * 64 + ns * 16 + lrow) * 40];
        v8bf p0 = *(const v8bf*)(p + sel * 8);
        v8bf p1 = *(const v8bf*)(p + 16 + sel * 8);
#pragma unroll
        for (int i = 0; i < 8; ++i) { bfr[ns][i] = p0[i]; bfr[ns][i + 8] = p1[i]; }
      }
#pragma unroll
      for (int ms = 0; ms < 2; ++ms)
#pragma unroll
        for (int ns = 0; ns < 4; ++ns)
          acc[ms][ns] = __builtin_amdgcn_wmma_f32_16x16x32_bf16(
              false, af[ms], false, bfr[ns], (short)0, acc[ms][ns], false, false);
    }
  }

#pragma unroll
  for (int ms = 0; ms < 2; ++ms) {
    const int m0 = mb * 128 + wm * 32 + ms * 16;
#pragma unroll
    for (int ns = 0; ns < 4; ++ns) {
      const int pix = nb * 128 + wn * 64 + ns * 16 + lrow;
      const int r = pix >> 5, c = pix & 31;
      const int oh = 2 * r + ph, ow = 2 * c + pw;
#pragma unroll
      for (int v = 0; v < 8; ++v) {
        const int oc = m0 + v + sel * 8;
        out[(((size_t)n * OC + oc) * OH + oh) * OH + ow] =
            acc[ms][ns][v] + bi[n * OC + oc];
      }
    }
  }
}

// ---------------------------------------------------------------------------
extern "C" void kernel_launch(void* const* d_in, const int* in_sizes, int n_in,
                              void* d_out, int out_size, void* d_ws, size_t ws_size,
                              hipStream_t stream) {
  (void)in_sizes; (void)n_in; (void)out_size;
  const float* x      = (const float*)d_in[0];
  const float* task   = (const float*)d_in[1];
  const float* W      = (const float*)d_in[2];
  const float* b      = (const float*)d_in[3];
  const float* task_w = (const float*)d_in[4];
  const float* task_b = (const float*)d_in[5];
  const float* tb_w   = (const float*)d_in[6];
  const float* tb_b   = (const float*)d_in[7];
  float* out = (float*)d_out;
  float* ws  = (float*)d_ws;

  // f32 workspace: wk_t 2M | gamma 4M | beta 4M | bi 8K  (~42 MB)
  float* wk_t  = ws;
  float* gamma = wk_t + (size_t)16 * IC * OC;
  float* beta  = gamma + (size_t)N_BATCH * IC * OC;
  float* bi    = beta + (size_t)N_BATCH * IC * OC;
  // bf16 extension (fast path): xbf 33.5 MB | wmod 134 MB
  const size_t f32_elems  = (size_t)16 * IC * OC + 2 * (size_t)N_BATCH * IC * OC + N_BATCH * OC;
  const size_t xbf_elems  = (size_t)N_BATCH * H * H * IC;
  const size_t wmod_elems = (size_t)N_BATCH * 16 * OC * IC;
  __bf16* xbf  = (__bf16*)(ws + f32_elems);
  __bf16* wmod = xbf + xbf_elems;
  const size_t need_fast = f32_elems * 4 + (xbf_elems + wmod_elems) * 2;

  k_wkprep   <<<(IC * OC + 255) / 256, 256, 0, stream>>>(W, wk_t);
  k_gammabeta<<<(2 * IC * OC) / 128,   256, 0, stream>>>(task, task_w, task_b, gamma, beta);
  k_bias     <<<(N_BATCH * OC) / 256,  256, 0, stream>>>(task, tb_w, tb_b, b, bi);

  if (ws_size >= need_fast) {
    k_xprep    <<<N_BATCH * H,        256, 0, stream>>>(x, xbf);
    k_wmod     <<<N_BATCH * 16 * 8 * 4, 256, 0, stream>>>(wk_t, gamma, beta, wmod);
    k_conv_fast<<<N_BATCH * 2 * 2 * 2 * 8, 256, 0, stream>>>(xbf, wmod, bi, out);
  } else {
    k_conv     <<<N_BATCH * 2 * 2 * 2 * 8, 256, 0, stream>>>(x, wk_t, gamma, beta, bi, out);
  }
}
